// MySeperableLayerNorm_47665547051246
// MI455X (gfx1250) — matrix operand order repaired and checked by reading, not compiled, not run on hardware
//
#include <hip/hip_runtime.h>

typedef __attribute__((ext_vector_type(16))) _Float16 v16h;
typedef __attribute__((ext_vector_type(8)))  float    v8f;
typedef __attribute__((ext_vector_type(4)))  float    v4f;

#define EPS 1e-6f

union AFrag { unsigned u[8]; v16h v; };
union CAcc  { float f[8];    v8f  v; };

__global__ __launch_bounds__(128)
void fused_sep_norm_mlp(const float* __restrict__ x,
                        const float* __restrict__ W1,
                        const float* __restrict__ b1,
                        const float* __restrict__ W2,
                        const float* __restrict__ b2,
                        const float* __restrict__ aff_w,
                        const float* __restrict__ aff_b,
                        float* __restrict__ out,
                        int n)
{
    // feats (f16, 64 rows x 224) -- reused as gate (f16) after GEMM1 consumed it
    __shared__ __align__(16) _Float16 sFeats[64 * 224];
    __shared__ __align__(16) _Float16 sW1B[28 * 512];   // 7 ksteps x 4 ntiles, B-layout
    __shared__ __align__(16) _Float16 sW2B[28 * 512];   // 2 ksteps x 14 ntiles, B-layout
    __shared__ __align__(16) _Float16 sH[64 * 64];      // hidden, K padded 56->64
    __shared__ float sRed[128];
    __shared__ float sRed2[128];
    __shared__ float sMean[64];
    __shared__ float sRnorm[64];
    __shared__ float sB1[56];
    __shared__ float sB2[224];
    __shared__ float sAffW[224];
    __shared__ float sAffB[128];

    const int tid     = threadIdx.x;
    const int rowBase = blockIdx.x * 64;

    // ---------------- Phase 0: weights -> LDS in WMMA B-layout (f16) -------------
    // B tile k order used here (contiguous, A matches): lane L: N = L%16, K = kstep*32 + (L>=16)*16 + i,
    // i = 0..15 packed halves within the lane's 8 VGPRs.
    for (int idx = tid; idx < 28 * 512; idx += 128) {
        int t = idx >> 9, rem = idx & 511, ln = rem >> 4, i = rem & 15;
        int ks = t >> 2, nt = t & 3;                 // GEMM1: 7 ksteps x 4 ntiles
        int K = ks * 32 + ((ln >> 4) << 4) + i;      // 0..223
        int N = nt * 16 + (ln & 15);                 // 0..63 (pad >=56)
        sW1B[idx] = (N < 56) ? (_Float16)W1[N * 224 + K] : (_Float16)0.f;
    }
    for (int idx = tid; idx < 28 * 512; idx += 128) {
        int t = idx >> 9, rem = idx & 511, ln = rem >> 4, i = rem & 15;
        int ks = t / 14, nt = t % 14;                // GEMM2: 2 ksteps x 14 ntiles
        int K = ks * 32 + ((ln >> 4) << 4) + i;      // 0..63 (pad >=56)
        int N = nt * 16 + (ln & 15);                 // 0..223
        sW2B[idx] = (K < 56) ? (_Float16)W2[N * 56 + K] : (_Float16)0.f;
    }
    if (tid < 56) sB1[tid] = b1[tid];
    for (int i = tid; i < 224; i += 128) { sB2[i] = b2[i]; sAffW[i] = aff_w[i]; }
    sAffB[tid] = aff_b[tid];

    // ---------------- Phase 1: per-row feature build (2 threads / row) -----------
    const int r  = tid >> 1;
    const int h2 = tid & 1;
    const int g0 = rowBase + r;
    const int g  = (g0 < n) ? g0 : (n - 1);          // clamp for safe loads
    const float* xr = x + (size_t)g * 480;

    float s = 0.f;
    {
        const v4f* p = (const v4f*)(xr + h2 * 64);
        #pragma unroll
        for (int i = 0; i < 16; i++) { v4f v = p[i]; s += v.x + v.y + v.z + v.w; }
    }
    sRed[tid] = s;
    __syncthreads();
    const float mean = (sRed[r * 2] + sRed[r * 2 + 1]) * (1.f / 128.f);

    float fsq = 0.f;
    {
        const float* p  = xr + h2 * 64;
        _Float16*    fo = &sFeats[r * 224 + h2 * 64];
        #pragma unroll 8
        for (int i = 0; i < 64; i++) {
            float sc = p[i] - mean;
            fo[i] = (_Float16)sc;
            fsq += sc * sc;
        }
    }
    if (h2 == 0) {
        #pragma unroll 4
        for (int ch = 0; ch < 48; ch++) {            // 3-groups, cols 128..271
            const float* p = xr + 128 + ch * 3;
            float ss = p[0]*p[0] + p[1]*p[1] + p[2]*p[2];
            float rms = sqrtf(ss * (1.f / 3.f) + EPS);
            sFeats[r * 224 + 128 + ch] = (_Float16)rms;
            fsq += rms * rms;
        }
    } else {
        #pragma unroll 4
        for (int ch = 48; ch < 64; ch++) {           // 3-groups, cols 272..319
            const float* p = xr + 128 + ch * 3;
            float ss = p[0]*p[0] + p[1]*p[1] + p[2]*p[2];
            float rms = sqrtf(ss * (1.f / 3.f) + EPS);
            sFeats[r * 224 + 128 + ch] = (_Float16)rms;
            fsq += rms * rms;
        }
        #pragma unroll 4
        for (int ch = 64; ch < 96; ch++) {           // 5-groups, cols 320..479
            const float* p = xr + 320 + (ch - 64) * 5;
            float ss = p[0]*p[0] + p[1]*p[1] + p[2]*p[2] + p[3]*p[3] + p[4]*p[4];
            float rms = sqrtf(ss * (1.f / 5.f) + EPS);
            sFeats[r * 224 + 128 + ch] = (_Float16)rms;
            fsq += rms * rms;
        }
    }
    sRed2[tid] = fsq;
    __syncthreads();
    {
        float tot = sRed2[r * 2] + sRed2[r * 2 + 1];
        float rn  = rsqrtf(tot * (1.f / 224.f) + EPS);
        if (h2 == 0) { sMean[r] = mean; sRnorm[r] = rn; }
    }
    __syncthreads();

    // ---------------- Phase 2: GEMM1 (feats @ W1^T) + silu, per wave -------------
    const int w    = tid >> 5;
    const int lane = tid & 31;
    const int mrow = lane & 15;
    const int hi   = lane >> 4;

    {
        CAcc acc[4];
        #pragma unroll
        for (int nt = 0; nt < 4; nt++)
            #pragma unroll
            for (int i = 0; i < 8; i++) acc[nt].f[i] = 0.f;

        const _Float16* frow = &sFeats[(w * 16 + mrow) * 224];
        #pragma unroll
        for (int ks = 0; ks < 7; ks++) {
            AFrag a;
            #pragma unroll
            for (int v = 0; v < 8; v++) {
                int K0 = ks * 32 + hi * 16 + (v << 1);
                a.u[v] = *(const unsigned*)(frow + K0);
            }
            #pragma unroll
            for (int nt = 0; nt < 4; nt++) {
                const v16h b = *(const v16h*)(sW1B + (ks * 4 + nt) * 512 + lane * 16);
                acc[nt].v = __builtin_amdgcn_wmma_f32_16x16x32_f16(
                    false, a.v, false, b, (short)0, acc[nt].v, false, false);
            }
        }
        // fold per-row rsqrt-norm into GEMM1 result, add b1, silu, pack to A source
        #pragma unroll
        for (int nt = 0; nt < 4; nt++) {
            #pragma unroll
            for (int rr = 0; rr < 8; rr++) {
                int M = rr + hi * 8;
                int grow = w * 16 + M;
                int j = nt * 16 + mrow;              // 0..63
                float z  = acc[nt].f[rr] * sRnorm[grow] + ((j < 56) ? sB1[j] : 0.f);
                float hv = z * (1.f / (1.f + __expf(-z)));
                sH[grow * 64 + j] = (j < 56) ? (_Float16)hv : (_Float16)0.f;
            }
        }
    }
    __syncthreads();

    // ---------------- Phase 3: GEMM2 (h @ W2^T) + sigmoid*aff_w -> gate ----------
    {
        AFrag a2[2];
        const _Float16* hrow = &sH[(w * 16 + mrow) * 64];
        #pragma unroll
        for (int ks = 0; ks < 2; ks++)
            #pragma unroll
            for (int v = 0; v < 8; v++) {
                int K0 = ks * 32 + hi * 16 + (v << 1);
                a2[ks].u[v] = *(const unsigned*)(hrow + K0);
            }

        #pragma unroll 2
        for (int nt = 0; nt < 14; nt++) {
            CAcc acc;
            #pragma unroll
            for (int i = 0; i < 8; i++) acc.f[i] = 0.f;
            {
                const v16h b0 = *(const v16h*)(sW2B + (0 * 14 + nt) * 512 + lane * 16);
                acc.v = __builtin_amdgcn_wmma_f32_16x16x32_f16(
                    false, a2[0].v, false, b0, (short)0, acc.v, false, false);
                const v16h b1t = *(const v16h*)(sW2B + (1 * 14 + nt) * 512 + lane * 16);
                acc.v = __builtin_amdgcn_wmma_f32_16x16x32_f16(
                    false, a2[1].v, false, b1t, (short)0, acc.v, false, false);
            }
            #pragma unroll
            for (int rr = 0; rr < 8; rr++) {
                int M = rr + hi * 8;
                int grow = w * 16 + M;
                int j = nt * 16 + mrow;              // 0..223
                float z  = acc.f[rr] + sB2[j];
                float gv = sAffW[j] * (1.f / (1.f + __expf(-z)));
                sFeats[grow * 224 + j] = (_Float16)gv;   // reuse feats buffer as gate
            }
        }
    }
    __syncthreads();

    // ---------------- Phase 4: gated output (coalesced float4, NT stores) --------
    for (int k2 = tid; k2 < 64 * 120; k2 += 128) {
        int row = k2 / 120;
        int c4  = k2 - row * 120;
        int gr  = rowBase + row;
        if (gr >= n) continue;
        const v4f xv = *(const v4f*)(x + (size_t)gr * 480 + c4 * 4);
        const _Float16* gate = &sFeats[row * 224];
        v4f ov;
        #pragma unroll
        for (int cc = 0; cc < 4; cc++) {
            int col = c4 * 4 + cc;
            float xc = xv[cc];
            float o;
            if (col < 128) {
                o = (xc - sMean[row]) * (float)gate[col] + sAffB[col];
            } else if (col < 320) {
                int ch = 128 + (col - 128) / 3;
                o = xc * (float)gate[ch];
            } else {
                int ch = 192 + (col - 320) / 5;
                o = xc * (float)gate[ch];
            }
            ov[cc] = o;
        }
        __builtin_nontemporal_store(ov, (v4f*)(out + (size_t)gr * 480 + c4 * 4));
    }
}

extern "C" void kernel_launch(void* const* d_in, const int* in_sizes, int n_in,
                              void* d_out, int out_size, void* d_ws, size_t ws_size,
                              hipStream_t stream) {
    (void)n_in; (void)out_size; (void)d_ws; (void)ws_size;
    const float* x     = (const float*)d_in[0];
    const float* W1    = (const float*)d_in[1];
    const float* b1    = (const float*)d_in[2];
    const float* W2    = (const float*)d_in[3];
    const float* b2    = (const float*)d_in[4];
    const float* aff_w = (const float*)d_in[5];
    const float* aff_b = (const float*)d_in[6];
    float* out = (float*)d_out;
    const int n = in_sizes[0] / 480;
    if (n <= 0) return;
    const int blocks = (n + 63) / 64;
    fused_sep_norm_mlp<<<dim3(blocks), dim3(128), 0, stream>>>(
        x, W1, b1, W2, b2, aff_w, aff_b, out, n);
}
